// MultiHeadAttention_21955872817720
// MI455X (gfx1250) — matrix operand order repaired and checked by reading, not compiled, not run on hardware
//
#include <hip/hip_runtime.h>
#include <hip/hip_bf16.h>

#define B_   2
#define S_   2048
#define D_   1024
#define H_   16
#define DK_  64
#define MTOT (B_ * S_)   // 4096

typedef __attribute__((ext_vector_type(16))) __bf16       v16bf;
typedef __attribute__((ext_vector_type(8)))  __bf16       v8bf;
typedef __attribute__((ext_vector_type(8)))  float        v8f;
typedef __attribute__((ext_vector_type(4)))  unsigned int u32x4;
typedef __attribute__((ext_vector_type(4)))  int          i32x4;
typedef __attribute__((ext_vector_type(8)))  int          i32x8;

static __device__ __forceinline__ v16bf cat8(v8bf lo, v8bf hi) {
  return __builtin_shufflevector(lo, hi, 0,1,2,3,4,5,6,7,8,9,10,11,12,13,14,15);
}

static __device__ __forceinline__ v8bf cvt8(float4 a, float4 b) {
  v8bf r;
  r[0] = (__bf16)a.x; r[1] = (__bf16)a.y; r[2] = (__bf16)a.z; r[3] = (__bf16)a.w;
  r[4] = (__bf16)b.x; r[5] = (__bf16)b.y; r[6] = (__bf16)b.z; r[7] = (__bf16)b.w;
  return r;
}

// A-matrix 16x32 bf16 fragment. Lane L: M = L&15, half = L>>4.
// element e -> k = (e&8)*2 + half*8 + (e&7). base = row ptr already offset by k0.
static __device__ __forceinline__ v16bf load_a_bf16(const __bf16* base, int lh) {
  v8bf lo = *(const v8bf*)(base + lh * 8);
  v8bf hi = *(const v8bf*)(base + 16 + lh * 8);
  return cat8(lo, hi);
}
// B-matrix 32x16 bf16 fragment. Lane L: N = L&15, half = L>>4; element e -> K = (e&8)*2 + half*8 + (e&7).
static __device__ __forceinline__ v16bf load_b_bf16(const __bf16* base, int lh) {
  v8bf lo = *(const v8bf*)(base + lh * 8);
  v8bf hi = *(const v8bf*)(base + 16 + lh * 8);
  return cat8(lo, hi);
}

static __device__ __forceinline__ v8f wmma_bf16(v16bf a, v16bf b, v8f c) {
  return __builtin_amdgcn_wmma_f32_16x16x32_bf16(false, a, false, b, (short)0, c, false, false);
}

// ---------------------------------------------------------------------------
// TDM: 2D tile load Global -> LDS. Descriptor per CDNA5 ISA.
// tile_d0 = contiguous elements per line, tile_d1 = #lines, stride_d0 = line stride
// (all in bf16 elements, data_size code 1 = 2 bytes).
// ---------------------------------------------------------------------------
static __device__ __forceinline__ void tdm_load_2d(const void* gptr, unsigned lds_off,
                                                   unsigned tile_d0, unsigned tile_d1,
                                                   unsigned stride_d0)
{
  unsigned long long ga = (unsigned long long)gptr;
  u32x4 g0;
  g0[0] = 1u;                                            // count=1, user descriptor
  g0[1] = lds_off;                                       // lds_addr (bytes)
  g0[2] = (unsigned)(ga & 0xFFFFFFFFu);                  // global_addr[31:0]
  g0[3] = (unsigned)((ga >> 32) & 0x01FFFFFFu) | (2u << 30);  // addr[56:32] | type=2
  i32x8 g1;
  g1[0] = (int)(1u << 16);                               // workgroup_mask=0, data_size=1 (2B)
  g1[1] = (int)(tile_d0 << 16);                          // tensor_dim0[15:0] = tile_d0
  g1[2] = (int)(tile_d1 << 16);                          // tensor_dim1[15:0] = tile_d1
  g1[3] = (int)(tile_d0 << 16);                          // tile_dim0
  g1[4] = (int)(tile_d1 & 0xFFFFu);                      // tile_dim1, tile_dim2=0 (2D)
  g1[5] = (int)stride_d0;                                // tensor_dim0_stride[31:0]
  g1[6] = 0;
  g1[7] = 0;
  i32x4 z4 = {0, 0, 0, 0};
#if __has_include(<hip/amd_detail/amd_gfx1250_TDM.h>)
  i32x8 z8 = {0, 0, 0, 0, 0, 0, 0, 0};
  __builtin_amdgcn_tensor_load_to_lds(g0, g1, z4, z4, z8, 0);
#else
  __builtin_amdgcn_tensor_load_to_lds(g0, g1, z4, z4, 0);
#endif
}

// ---------------------------------------------------------------------------
// Kernel 0: one-shot f32 -> bf16 conversion (8 elements / thread).
// ---------------------------------------------------------------------------
__global__ __launch_bounds__(256) void cvt_bf16_kernel(const float* __restrict__ src,
                                                       __bf16* __restrict__ dst, int n)
{
  const int i = (blockIdx.x * 256 + threadIdx.x) * 8;
  if (i < n) {
    const float4* p = (const float4*)(src + i);
    *(v8bf*)(dst + i) = cvt8(p[0], p[1]);
  }
}

// ---------------------------------------------------------------------------
// Kernel 1: fused QKV projection (all-bf16 operands).  out = x @ W^T
// blockIdx.z: 0 -> Q ([B][H][S][DK]), 1 -> K (same), 2 -> V ([B][H][DK][S])
// One wave = 32x32 output tile, blockDim (32,4).
// ---------------------------------------------------------------------------
__global__ __launch_bounds__(128) void qkv_proj_kernel(
    const __bf16* __restrict__ x,
    const __bf16* __restrict__ Wq, const __bf16* __restrict__ Wk, const __bf16* __restrict__ Wv,
    __bf16* __restrict__ q_ws, __bf16* __restrict__ k_ws, __bf16* __restrict__ vt_ws)
{
  const int lane  = threadIdx.x;
  const int wave  = threadIdx.y;
  const int lh    = lane >> 4;
  const int lan16 = lane & 15;

  const int tiles_n = D_ / 32;                 // 32
  const int tile = blockIdx.x * 4 + wave;      // 0..4095
  const int tm = tile / tiles_n, tn = tile % tiles_n;
  const int m0 = tm * 32, n0 = tn * 32;

  const __bf16* W;
  if      (blockIdx.z == 0) W = Wq;
  else if (blockIdx.z == 1) W = Wk;
  else                      W = Wv;

  v8f acc[2][2] = {};
  const __bf16* ar0 = x + (size_t)(m0 + lan16)      * D_;
  const __bf16* ar1 = x + (size_t)(m0 + 16 + lan16) * D_;
  const __bf16* br0 = W + (size_t)(n0 + lan16)      * D_;
  const __bf16* br1 = W + (size_t)(n0 + 16 + lan16) * D_;

  for (int k0 = 0; k0 < D_; k0 += 32) {
    __builtin_prefetch(ar0 + k0 + 128, 0, 1);
    __builtin_prefetch(br0 + k0 + 128, 0, 1);
    v16bf a0 = load_a_bf16(ar0 + k0, lh);
    v16bf a1 = load_a_bf16(ar1 + k0, lh);
    v16bf b0 = load_b_bf16(br0 + k0, lh);
    v16bf b1 = load_b_bf16(br1 + k0, lh);
    acc[0][0] = wmma_bf16(a0, b0, acc[0][0]);
    acc[0][1] = wmma_bf16(a0, b1, acc[0][1]);
    acc[1][0] = wmma_bf16(a1, b0, acc[1][0]);
    acc[1][1] = wmma_bf16(a1, b1, acc[1][1]);
  }

  __bf16* dst = (blockIdx.z == 0) ? q_ws : (blockIdx.z == 1 ? k_ws : vt_ws);
  const bool vmode = (blockIdx.z == 2);
  #pragma unroll
  for (int mt = 0; mt < 2; ++mt)
    #pragma unroll
    for (int nt = 0; nt < 2; ++nt) {
      const int n = n0 + nt * 16 + lan16;
      const int h = n >> 6, dk = n & 63;
      #pragma unroll
      for (int r = 0; r < 8; ++r) {
        const int m = m0 + mt * 16 + lh * 8 + r;
        const int b = m >> 11, s = m & (S_ - 1);
        size_t idx;
        if (!vmode) idx = ((size_t)(b * H_ + h) * S_  + s)  * DK_ + dk;   // [B][H][S][DK]
        else        idx = ((size_t)(b * H_ + h) * DK_ + dk) * S_  + s;    // [B][H][DK][S]
        dst[idx] = (__bf16)acc[mt][nt][r];
      }
    }
}

// ---------------------------------------------------------------------------
// Kernel 2: causal flash attention.
// One block = one (b,h), 4 waves = 4 consecutive 32-row q-tiles (128 rows).
// Shared K (32x64) and V^T (64x32) tiles are TDM-DMA'd into double-buffered LDS;
// wave 0 drives the TDM pipeline (s_wait_tensorcnt), barriers publish buffers.
// ---------------------------------------------------------------------------
__global__ __launch_bounds__(128) void attn_kernel(
    const __bf16* __restrict__ q_ws, const __bf16* __restrict__ k_ws,
    const __bf16* __restrict__ vt_ws, __bf16* __restrict__ attn_ws)
{
  __shared__ __bf16 lds_k[2][32 * 64];      // [buf][j_local][dk]
  __shared__ __bf16 lds_v[2][64 * 32];      // [buf][dk][j_local]
  __shared__ __bf16 lds_p[4][2][16 * 32];   // per-wave P staging (C-layout -> A-layout)

  const int lane  = threadIdx.x;
  const int wave  = threadIdx.y;
  const int lh    = lane >> 4;
  const int lan16 = lane & 15;

  const int bh  = blockIdx.x >> 4;          // b*H + h   (16 q-blocks per head)
  const int qb  = blockIdx.x & 15;
  const int m0b = qb * 128;                 // block's first query row
  const int m0  = m0b + wave * 32;          // this wave's q-tile

  const __bf16* qh = q_ws  + (size_t)bh * S_ * DK_;
  const __bf16* kh = k_ws  + (size_t)bh * S_ * DK_;
  const __bf16* vh = vt_ws + (size_t)bh * DK_ * S_;

  // Q fragments: 2 m-tiles x 2 k-steps, resident for the whole j-loop.
  v16bf qf[2][2];
  #pragma unroll
  for (int mt = 0; mt < 2; ++mt) {
    const __bf16* qrow = qh + (size_t)(m0 + mt * 16 + lan16) * DK_;
    qf[mt][0] = load_a_bf16(qrow + 0,  lh);
    qf[mt][1] = load_a_bf16(qrow + 32, lh);
  }

  v8f o_acc[2][4] = {};
  float mstate[2][8], lstate[2][8];
  #pragma unroll
  for (int mt = 0; mt < 2; ++mt)
    #pragma unroll
    for (int r = 0; r < 8; ++r) { mstate[mt][r] = -3.0e38f; lstate[mt][r] = 0.f; }

  const float scale = 0.125f;               // 1/sqrt(64)
  const int njt = (m0b + 96) / 32 + 1;      // j-tiles needed by the whole block

  // prologue: DMA tile 0 into buffer 0
  if (wave == 0) {
    tdm_load_2d(kh,      (unsigned)(size_t)&lds_k[0][0], 64, 32, 64);
    tdm_load_2d(vh,      (unsigned)(size_t)&lds_v[0][0], 32, 64, S_);
  }

  for (int jt = 0; jt < njt; ++jt) {
    const int buf = jt & 1;
    const int j0  = jt * 32;

    if (wave == 0) __builtin_amdgcn_s_wait_tensorcnt(0);
    __syncthreads();                         // buffer `buf` ready; prev compute done
    if (wave == 0 && jt + 1 < njt) {
      const int jn = (jt + 1) * 32;
      tdm_load_2d(kh + (size_t)jn * DK_, (unsigned)(size_t)&lds_k[buf ^ 1][0], 64, 32, 64);
      tdm_load_2d(vh + jn,               (unsigned)(size_t)&lds_v[buf ^ 1][0], 32, 64, S_);
    }

    if (j0 <= m0) {
      // ---- scores = Q @ K^T (K from LDS) --------------------------------
      v16bf kb[2][2];
      #pragma unroll
      for (int nt = 0; nt < 2; ++nt) {
        const __bf16* krow = &lds_k[buf][(nt * 16 + lan16) * 64];
        kb[nt][0] = load_b_bf16(krow + 0,  lh);
        kb[nt][1] = load_b_bf16(krow + 32, lh);
      }
      v8f sc[2][2] = {};
      #pragma unroll
      for (int mt = 0; mt < 2; ++mt)
        #pragma unroll
        for (int nt = 0; nt < 2; ++nt) {
          sc[mt][nt] = wmma_bf16(qf[mt][0], kb[nt][0], sc[mt][nt]);
          sc[mt][nt] = wmma_bf16(qf[mt][1], kb[nt][1], sc[mt][nt]);
        }

      const bool diag = (j0 == m0);          // only diagonal tile crosses causal edge
      #pragma unroll
      for (int mt = 0; mt < 2; ++mt)
        #pragma unroll
        for (int nt = 0; nt < 2; ++nt)
          #pragma unroll
          for (int r = 0; r < 8; ++r) {
            float v = sc[mt][nt][r] * scale;
            if (diag) {
              const int qrow = m0 + mt * 16 + lh * 8 + r;
              const int kcol = j0 + nt * 16 + lan16;
              if (kcol > qrow) v = -3.0e38f;
            }
            sc[mt][nt][r] = v;
          }

      // ---- online softmax ------------------------------------------------
      #pragma unroll
      for (int mt = 0; mt < 2; ++mt) {
        #pragma unroll
        for (int r = 0; r < 8; ++r) {
          float v = fmaxf(sc[mt][0][r], sc[mt][1][r]);
          v = fmaxf(v, __shfl_xor(v, 1));
          v = fmaxf(v, __shfl_xor(v, 2));
          v = fmaxf(v, __shfl_xor(v, 4));
          v = fmaxf(v, __shfl_xor(v, 8));
          const float newm = fmaxf(mstate[mt][r], v);
          const float corr = __expf(mstate[mt][r] - newm);
          mstate[mt][r] = newm;
          const float p0 = __expf(sc[mt][0][r] - newm);
          const float p1 = __expf(sc[mt][1][r] - newm);
          sc[mt][0][r] = p0; sc[mt][1][r] = p1;
          float rs = p0 + p1;
          rs += __shfl_xor(rs, 1);
          rs += __shfl_xor(rs, 2);
          rs += __shfl_xor(rs, 4);
          rs += __shfl_xor(rs, 8);
          lstate[mt][r] = lstate[mt][r] * corr + rs;
          #pragma unroll
          for (int dt = 0; dt < 4; ++dt) o_acc[mt][dt][r] *= corr;
        }
        __bf16* lp = lds_p[wave][mt];
        #pragma unroll
        for (int nt = 0; nt < 2; ++nt)
          #pragma unroll
          for (int r = 0; r < 8; ++r)
            lp[(lh * 8 + r) * 32 + nt * 16 + lan16] = (__bf16)sc[mt][nt][r];
      }

      // ---- O += P @ V (V from LDS) --------------------------------------
      v16bf vb[4];
      #pragma unroll
      for (int dt = 0; dt < 4; ++dt)
        vb[dt] = load_b_bf16(&lds_v[buf][(dt * 16 + lan16) * 32], lh);
      #pragma unroll
      for (int mt = 0; mt < 2; ++mt) {
        v16bf pf = load_a_bf16(lds_p[wave][mt] + lan16 * 32, lh);
        #pragma unroll
        for (int dt = 0; dt < 4; ++dt)
          o_acc[mt][dt] = wmma_bf16(pf, vb[dt], o_acc[mt][dt]);
      }
    }
    __syncthreads();                         // all readers done before buf is reused
  }

  // ---- normalize + merge heads: attn_ws[B][S][D] bf16 ----------------------
  const int b = bh >> 4, h = bh & 15;
  #pragma unroll
  for (int mt = 0; mt < 2; ++mt)
    #pragma unroll
    for (int r = 0; r < 8; ++r) {
      const float inv = 1.0f / lstate[mt][r];
      const int s = m0 + mt * 16 + lh * 8 + r;
      #pragma unroll
      for (int dt = 0; dt < 4; ++dt) {
        const int dk = dt * 16 + lan16;
        attn_ws[((size_t)(b * S_ + s)) * D_ + h * DK_ + dk] = (__bf16)(o_acc[mt][dt][r] * inv);
      }
    }
}

// ---------------------------------------------------------------------------
// Kernel 3: output projection (all-bf16 operands), f32 result.
// ---------------------------------------------------------------------------
__global__ __launch_bounds__(128) void out_proj_kernel(
    const __bf16* __restrict__ attn, const __bf16* __restrict__ Wo, float* __restrict__ out)
{
  const int lane  = threadIdx.x;
  const int wave  = threadIdx.y;
  const int lh    = lane >> 4;
  const int lan16 = lane & 15;

  const int tiles_n = D_ / 32;
  const int tile = blockIdx.x * 4 + wave;
  const int tm = tile / tiles_n, tn = tile % tiles_n;
  const int m0 = tm * 32, n0 = tn * 32;

  v8f acc[2][2] = {};
  const __bf16* ar0 = attn + (size_t)(m0 + lan16)      * D_;
  const __bf16* ar1 = attn + (size_t)(m0 + 16 + lan16) * D_;
  const __bf16* br0 = Wo   + (size_t)(n0 + lan16)      * D_;
  const __bf16* br1 = Wo   + (size_t)(n0 + 16 + lan16) * D_;

  for (int k0 = 0; k0 < D_; k0 += 32) {
    __builtin_prefetch(br0 + k0 + 128, 0, 1);
    v16bf a0 = load_a_bf16(ar0 + k0, lh);
    v16bf a1 = load_a_bf16(ar1 + k0, lh);
    v16bf b0 = load_b_bf16(br0 + k0, lh);
    v16bf b1 = load_b_bf16(br1 + k0, lh);
    acc[0][0] = wmma_bf16(a0, b0, acc[0][0]);
    acc[0][1] = wmma_bf16(a0, b1, acc[0][1]);
    acc[1][0] = wmma_bf16(a1, b0, acc[1][0]);
    acc[1][1] = wmma_bf16(a1, b1, acc[1][1]);
  }

  #pragma unroll
  for (int mt = 0; mt < 2; ++mt)
    #pragma unroll
    for (int nt = 0; nt < 2; ++nt) {
      const int n = n0 + nt * 16 + lan16;
      #pragma unroll
      for (int r = 0; r < 8; ++r) {
        const int m = m0 + mt * 16 + lh * 8 + r;
        out[(size_t)m * D_ + n] = acc[mt][nt][r];
      }
    }
}

// ---------------------------------------------------------------------------
extern "C" void kernel_launch(void* const* d_in, const int* in_sizes, int n_in,
                              void* d_out, int out_size, void* d_ws, size_t ws_size,
                              hipStream_t stream) {
  const float* x  = (const float*)d_in[0];
  const float* Wq = (const float*)d_in[1];
  const float* Wk = (const float*)d_in[2];
  const float* Wv = (const float*)d_in[3];
  const float* Wo = (const float*)d_in[4];
  float* out = (float*)d_out;

  const size_t act = (size_t)MTOT * D_ * sizeof(__bf16);   // 8 MB
  const size_t wgt = (size_t)D_ * D_ * sizeof(__bf16);     // 2 MB
  char* ws = (char*)d_ws;
  __bf16* q_ws    = (__bf16*)(ws);
  __bf16* k_ws    = (__bf16*)(ws + act);
  __bf16* vt_ws   = (__bf16*)(ws + 2 * act);
  __bf16* attn_ws = (__bf16*)(ws + 3 * act);
  __bf16* x_bf    = (__bf16*)(ws + 4 * act);
  __bf16* wq_bf   = (__bf16*)(ws + 5 * act);
  __bf16* wk_bf   = (__bf16*)(ws + 5 * act + wgt);
  __bf16* wv_bf   = (__bf16*)(ws + 5 * act + 2 * wgt);
  __bf16* wo_bf   = (__bf16*)(ws + 5 * act + 3 * wgt);

  dim3 blk(32, 4);
  // one-shot bf16 conversions (x: 4M elems, each W: 1M elems)
  cvt_bf16_kernel<<<dim3((MTOT * D_) / (8 * 256)), 256, 0, stream>>>(x,  x_bf,  MTOT * D_);
  cvt_bf16_kernel<<<dim3((D_ * D_) / (8 * 256)),   256, 0, stream>>>(Wq, wq_bf, D_ * D_);
  cvt_bf16_kernel<<<dim3((D_ * D_) / (8 * 256)),   256, 0, stream>>>(Wk, wk_bf, D_ * D_);
  cvt_bf16_kernel<<<dim3((D_ * D_) / (8 * 256)),   256, 0, stream>>>(Wv, wv_bf, D_ * D_);
  cvt_bf16_kernel<<<dim3((D_ * D_) / (8 * 256)),   256, 0, stream>>>(Wo, wo_bf, D_ * D_);

  qkv_proj_kernel<<<dim3(1024, 1, 3), blk, 0, stream>>>(x_bf, wq_bf, wk_bf, wv_bf,
                                                        q_ws, k_ws, vt_ws);
  // B*H * (S/128) = 512 blocks, each: 4 q-tiles of one head, TDM-fed K/V
  attn_kernel<<<dim3(512), blk, 0, stream>>>(q_ws, k_ws, vt_ws, attn_ws);
  out_proj_kernel<<<dim3(1024), blk, 0, stream>>>(attn_ws, wo_bf, out);
}
